// MyModel_34677565947981
// MI455X (gfx1250) — compile-verified, matrix operand order checked by reading, not run on hardware
//
#include <hip/hip_runtime.h>
#include <math.h>

// ---------------------------------------------------------------------------
// Problem constants (fixed by the reference's setup_inputs()):
#define SEQ     64
#define NNODES  4096
#define NEDGES  65536
#define INCH    128
#define H2      512      // conv1 out = 2*HID
#define OUTF    256      // conv2 out = HID = OUT
// CPC constants for SEQ=64, timestamp=4, n_samples=8:
//   neg_dist = 64//6 = 10 ; end = 64-8-10-4+2 = 44 ; start = 64//8 = 8
#define CPC_START 8
#define CPC_CNT   36
#define CPC_TS    4
#define CPC_NS    8
#define CPC_NEG   10
#define HID_LEN   44     // t_samples[-1]+1
#define MP_CHUNKS 32     // max-pool stage-1 blocks (128 nodes each)

typedef __attribute__((ext_vector_type(16))) __bf16 v16bf;
typedef __attribute__((ext_vector_type(8)))  float  v8f;

// ---------------------------------------------------------------------------
// Utility kernels
__global__ void fill0_k(float* __restrict__ p, int n) {
    int i = blockIdx.x * blockDim.x + threadIdx.x;
    if (i < n) p[i] = 0.0f;
}

__global__ void zero_out2_k(float* __restrict__ o) {
    if (threadIdx.x < 2) o[threadIdx.x] = 0.0f;
}

// ---------------------------------------------------------------------------
// Pre-pack f32 weights [K,N] into WMMA B-fragment order (bf16):
//   out[(((kTile*nTiles + nTile)*32 + lane)*16 + e]
//     = W[kTile*32 + 8*(lane/16) + e + 8*(e>=8)][nTile*16 + lane%16]
// so each lane's 16 fragment elements are 32 contiguous bytes.
__global__ void pack_b_k(const float* __restrict__ W, __bf16* __restrict__ out,
                         int K, int N) {
    int i = blockIdx.x * blockDim.x + threadIdx.x;
    if (i >= K * N) return;
    const int e     = i & 15;
    const int lane  = (i >> 4) & 31;
    const int tile  = i >> 9;
    const int nT    = N >> 4;
    const int kTile = tile / nT;
    const int nTile = tile - kTile * nT;
    const int kh = lane >> 4, rc = lane & 15;
    const int kk = kTile * 32 + 8 * kh + e + ((e >= 8) ? 8 : 0);
    out[i] = (__bf16)W[(size_t)kk * N + nTile * 16 + rc];
}

// ---------------------------------------------------------------------------
// Graph aggregation: degree + scatter-add (mean divide fused downstream)
__global__ void degree_k(const int* __restrict__ dst, float* __restrict__ deg, int E) {
    int e = blockIdx.x * blockDim.x + threadIdx.x;
    if (e < E) atomicAdd(&deg[dst[e]], 1.0f);
}

// One block per edge; threads cover channels (coalesced row read + atomic row add)
__global__ void scatter_add_k(const float* __restrict__ x, const int* __restrict__ src,
                              const int* __restrict__ dst, float* __restrict__ agg, int C) {
    int e = blockIdx.x;
    int s = src[e], d = dst[e];
    const float* xs = x + (size_t)s * C;
    float*       ad = agg + (size_t)d * C;
    __builtin_prefetch(xs, 0, 1);        // global_prefetch_b8
    for (int c = threadIdx.x; c < C; c += blockDim.x)
        atomicAdd(&ad[c], xs[c]);
}

// ---------------------------------------------------------------------------
// Shared WMMA helpers (fragment layout per CDNA5 ISA 7.12.2, wave32):
//   A: lane holds row M=lane%16; element e holds K = 8*(lane/16) + e + 8*(e>=8)
//   B: lane holds col N=lane%16; same K mapping (pre-packed -> contiguous 32 B)
//   C/D: lane holds col N=lane%16; VGPR v holds row M = v + 8*(lane/16)
__device__ __forceinline__ v16bf load_a_frag(const float* __restrict__ ap,
                                             int k0, int kh, float scale) {
    const float4 p0 = *(const float4*)(ap + k0 + 8 * kh);
    const float4 p1 = *(const float4*)(ap + k0 + 8 * kh + 4);
    const float4 p2 = *(const float4*)(ap + k0 + 16 + 8 * kh);
    const float4 p3 = *(const float4*)(ap + k0 + 16 + 8 * kh + 4);
    v16bf a;
    a[0]  = (__bf16)(p0.x * scale); a[1]  = (__bf16)(p0.y * scale);
    a[2]  = (__bf16)(p0.z * scale); a[3]  = (__bf16)(p0.w * scale);
    a[4]  = (__bf16)(p1.x * scale); a[5]  = (__bf16)(p1.y * scale);
    a[6]  = (__bf16)(p1.z * scale); a[7]  = (__bf16)(p1.w * scale);
    a[8]  = (__bf16)(p2.x * scale); a[9]  = (__bf16)(p2.y * scale);
    a[10] = (__bf16)(p2.z * scale); a[11] = (__bf16)(p2.w * scale);
    a[12] = (__bf16)(p3.x * scale); a[13] = (__bf16)(p3.y * scale);
    a[14] = (__bf16)(p3.z * scale); a[15] = (__bf16)(p3.w * scale);
    return a;
}

// ---------------------------------------------------------------------------
// conv1 GEMM: C[M,N] = (A1/deg)@B1 + A2@B2 + bias, ReLU
__global__ __launch_bounds__(256)
void gemm_dual_wmma_k(const float* __restrict__ A1, const float* __restrict__ A2,
                      const __bf16* __restrict__ B1p, const __bf16* __restrict__ B2p,
                      const float* __restrict__ deg, const float* __restrict__ bias,
                      float* __restrict__ C, int M, int N, int K) {
    const int lane    = threadIdx.x & 31;
    const int wave    = threadIdx.x >> 5;
    const int nTiles  = N >> 4;
    const int colBase = blockIdx.x * 16;
    const int rowBase = (blockIdx.y * 8 + wave) * 16;
    const int kh = lane >> 4, rc = lane & 15;
    const int row = rowBase + rc;

    const float inv = 1.0f / fmaxf(deg[row], 1.0f);  // fused mean for A1
    const float* a1p = A1 + (size_t)row * K;
    const float* a2p = A2 + (size_t)row * K;
    const size_t bstride = (size_t)nTiles * 32 * 16;
    const __bf16* b1p = B1p + ((size_t)blockIdx.x * 32 + lane) * 16;
    const __bf16* b2p = B2p + ((size_t)blockIdx.x * 32 + lane) * 16;

    v8f acc = {};
    for (int k0 = 0, kt = 0; k0 < K; k0 += 32, ++kt) {
        const v16bf a1 = load_a_frag(a1p, k0, kh, inv);
        const v16bf a2 = load_a_frag(a2p, k0, kh, 1.0f);
        const v16bf b1 = *(const v16bf*)(b1p + (size_t)kt * bstride);
        const v16bf b2 = *(const v16bf*)(b2p + (size_t)kt * bstride);
        acc = __builtin_amdgcn_wmma_f32_16x16x32_bf16(false, a1, false, b1,
                                                      (short)0, acc, false, false);
        acc = __builtin_amdgcn_wmma_f32_16x16x32_bf16(false, a2, false, b2,
                                                      (short)0, acc, false, false);
    }

    const float bv = bias[colBase + rc];
#pragma unroll
    for (int v = 0; v < 8; ++v) {
        const int r = rowBase + v + 8 * kh;
        C[(size_t)r * N + colBase + rc] = fmaxf(acc[v] + bv, 0.0f);  // ReLU
    }
}

// ---------------------------------------------------------------------------
// Pre-aggregation GEMM for conv2: Y[M,N] = A@Bp  (no bias/activation).
// Mean-aggregation commutes with this right-multiply, so scattering Y (256 ch)
// instead of h1 (512 ch) halves the conv2 gather/atomic traffic.
__global__ __launch_bounds__(256)
void gemm_single_wmma_k(const float* __restrict__ A, const __bf16* __restrict__ Bp,
                        float* __restrict__ C, int M, int N, int K) {
    const int lane    = threadIdx.x & 31;
    const int wave    = threadIdx.x >> 5;
    const int nTiles  = N >> 4;
    const int colBase = blockIdx.x * 16;
    const int rowBase = (blockIdx.y * 8 + wave) * 16;
    const int kh = lane >> 4, rc = lane & 15;

    const float* ap = A + (size_t)(rowBase + rc) * K;
    const size_t bstride = (size_t)nTiles * 32 * 16;
    const __bf16* bp = Bp + ((size_t)blockIdx.x * 32 + lane) * 16;

    v8f acc = {};
    for (int k0 = 0, kt = 0; k0 < K; k0 += 32, ++kt) {
        const v16bf a = load_a_frag(ap, k0, kh, 1.0f);
        const v16bf b = *(const v16bf*)(bp + (size_t)kt * bstride);
        acc = __builtin_amdgcn_wmma_f32_16x16x32_bf16(false, a, false, b,
                                                      (short)0, acc, false, false);
    }
#pragma unroll
    for (int v = 0; v < 8; ++v) {
        const int r = rowBase + v + 8 * kh;
        C[(size_t)r * N + colBase + rc] = acc[v];
    }
}

// ---------------------------------------------------------------------------
// conv2 final: C[M,N] = AGG/deg + A2@B2 + bias  (no activation)
__global__ __launch_bounds__(256)
void gemm_addagg_wmma_k(const float* __restrict__ AGG, const float* __restrict__ A2,
                        const __bf16* __restrict__ B2p, const float* __restrict__ deg,
                        const float* __restrict__ bias, float* __restrict__ C,
                        int M, int N, int K) {
    const int lane    = threadIdx.x & 31;
    const int wave    = threadIdx.x >> 5;
    const int nTiles  = N >> 4;
    const int colBase = blockIdx.x * 16;
    const int rowBase = (blockIdx.y * 8 + wave) * 16;
    const int kh = lane >> 4, rc = lane & 15;

    const float* a2p = A2 + (size_t)(rowBase + rc) * K;
    const size_t bstride = (size_t)nTiles * 32 * 16;
    const __bf16* b2p = B2p + ((size_t)blockIdx.x * 32 + lane) * 16;

    v8f acc = {};
    for (int k0 = 0, kt = 0; k0 < K; k0 += 32, ++kt) {
        const v16bf a = load_a_frag(a2p, k0, kh, 1.0f);
        const v16bf b = *(const v16bf*)(b2p + (size_t)kt * bstride);
        acc = __builtin_amdgcn_wmma_f32_16x16x32_bf16(false, a, false, b,
                                                      (short)0, acc, false, false);
    }

    const float bv = bias[colBase + rc];
#pragma unroll
    for (int v = 0; v < 8; ++v) {
        const int r = rowBase + v + 8 * kh;
        const float inv = 1.0f / fmaxf(deg[r], 1.0f);
        const size_t idx = (size_t)r * N + colBase + rc;
        C[idx] = acc[v] + AGG[idx] * inv + bv;
    }
}

// ---------------------------------------------------------------------------
// Global max-pool over nodes, two-stage deterministic reduction
__global__ void maxpool_part_k(const float* __restrict__ h, float* __restrict__ part,
                               int nodesPerBlk) {
    const int f = threadIdx.x;             // 256 features, coalesced
    const int n0 = blockIdx.x * nodesPerBlk;
    float m = -INFINITY;
    for (int n = n0; n < n0 + nodesPerBlk; ++n)
        m = fmaxf(m, h[(size_t)n * OUTF + f]);
    part[blockIdx.x * OUTF + f] = m;
}

__global__ void maxpool_fin_k(const float* __restrict__ part, float* __restrict__ z,
                              int chunks) {
    const int f = threadIdx.x;
    float m = -INFINITY;
    for (int c0 = 0; c0 < chunks; ++c0) m = fmaxf(m, part[c0 * OUTF + f]);
    z[f] = m;
}

// ---------------------------------------------------------------------------
// Single GRU-cell step per sequence element (h0 = 0)
__global__ void gru_k(const float* __restrict__ z, const float* __restrict__ Wih,
                      const float* __restrict__ bih, const float* __restrict__ bhh,
                      float* __restrict__ c) {
    __shared__ float zs[OUTF];
    const int t = blockIdx.x, f = threadIdx.x;
    zs[f] = z[(size_t)t * OUTF + f];
    __syncthreads();
    float g0 = bih[f], g1 = bih[OUTF + f], g2 = bih[2 * OUTF + f];
    for (int k = 0; k < OUTF; ++k) {
        const float zk = zs[k];
        g0 += Wih[(size_t)f * OUTF + k] * zk;
        g1 += Wih[(size_t)(OUTF + f) * OUTF + k] * zk;
        g2 += Wih[(size_t)(2 * OUTF + f) * OUTF + k] * zk;
    }
    const float r = 1.0f / (1.0f + __expf(-(g0 + bhh[f])));
    const float u = 1.0f / (1.0f + __expf(-(g1 + bhh[OUTF + f])));
    const float n = tanhf(g2 + r * bhh[2 * OUTF + f]);
    c[(size_t)t * OUTF + f] = (1.0f - u) * n;
}

// ---------------------------------------------------------------------------
// CPC: cosine scores, log-softmax over n_samples, accumulate nce & acc.
__global__ void cpc_k(const float* __restrict__ z, const float* __restrict__ c,
                      float* __restrict__ out) {
    const int tIdx = blockIdx.x;            // 0..35
    const int i    = blockIdx.y;            // 0..3
    const int ts   = CPC_START + tIdx;
    const int lane = threadIdx.x & 31;
    const int j    = threadIdx.x >> 5;      // sample index 0..7
    // off[i][0]=i+1 ; off[i][j>=1]=i+1+neg_dist+j-1 = i+j+10
    const int zi   = ts + ((j == 0) ? (i + 1) : (i + j + CPC_NEG));
    const float* enc = z + (size_t)zi * OUTF;
    const float* ct  = c + (size_t)ts * OUTF;

    float dot = 0.0f, ne = 0.0f, nc = 0.0f;
    for (int e = lane; e < OUTF; e += 32) {
        const float ev = enc[e], cv = ct[e];
        dot += ev * cv; ne += ev * ev; nc += cv * cv;
    }
    for (int off = 16; off; off >>= 1) {    // wave32 reduction
        dot += __shfl_xor(dot, off, 32);
        ne  += __shfl_xor(ne,  off, 32);
        nc  += __shfl_xor(nc,  off, 32);
    }
    __shared__ float tot[CPC_NS];
    if (lane == 0) {
        const float den = fmaxf(sqrtf(ne), 1e-8f) * fmaxf(sqrtf(nc), 1e-8f);
        tot[j] = dot / den;
    }
    __syncthreads();
    if (threadIdx.x == 0) {
        float m = tot[0];
        for (int q = 1; q < CPC_NS; ++q) m = fmaxf(m, tot[q]);
        float s = 0.0f;
        for (int q = 0; q < CPC_NS; ++q) s += __expf(tot[q] - m);
        const float nce = (m + __logf(s)) - tot[0];   // -log_softmax[...,0]
        bool first = true;                            // argmax==0 (ties -> 0)
        for (int q = 1; q < CPC_NS; ++q) if (tot[q] > tot[0]) first = false;
        const float inv = 1.0f / (float)(CPC_CNT * CPC_TS);
        atomicAdd(&out[0], nce * inv);
        atomicAdd(&out[1], (first ? 1.0f : 0.0f) * inv);
    }
}

__global__ void copy_hidden_k(const float* __restrict__ c, float* __restrict__ out) {
    int i = blockIdx.x * blockDim.x + threadIdx.x;
    if (i < HID_LEN * OUTF) out[2 + i] = c[i];
}

// ---------------------------------------------------------------------------
extern "C" void kernel_launch(void* const* d_in, const int* in_sizes, int n_in,
                              void* d_out, int out_size, void* d_ws, size_t ws_size,
                              hipStream_t stream) {
    const float* x   = (const float*)d_in[0];   // [SEQ, NNODES, INCH]
    const int*   ei  = (const int*)  d_in[1];   // [SEQ, 2, NEDGES]
    const float* W1l = (const float*)d_in[2];   // [128, 512]
    const float* b1l = (const float*)d_in[3];   // [512]
    const float* W1r = (const float*)d_in[4];   // [128, 512]
    const float* W2l = (const float*)d_in[5];   // [512, 256]
    const float* b2l = (const float*)d_in[6];   // [256]
    const float* W2r = (const float*)d_in[7];   // [512, 256]
    const float* Wih = (const float*)d_in[8];   // [768, 256]
    const float* bih = (const float*)d_in[10];  // [768]
    const float* bhh = (const float*)d_in[11];  // [768]
    float* out = (float*)d_out;                 // [1 + 1 + 44*256]

    // ---- carve workspace (256 B aligned) ----
    char* w = (char*)d_ws;
    auto take = [&](size_t bytes) {
        char* p = w;
        w += (bytes + 255) & ~(size_t)255;
        return p;
    };
    __bf16* W1l_b = (__bf16*)take((size_t)INCH * H2 * 2);
    __bf16* W1r_b = (__bf16*)take((size_t)INCH * H2 * 2);
    __bf16* W2l_b = (__bf16*)take((size_t)H2 * OUTF * 2);
    __bf16* W2r_b = (__bf16*)take((size_t)H2 * OUTF * 2);
    float* agg1  = (float*)take((size_t)NNODES * INCH * 4);
    float* deg   = (float*)take((size_t)NNODES * 4);
    float* h1    = (float*)take((size_t)NNODES * H2 * 4);
    float* y1    = (float*)take((size_t)NNODES * OUTF * 4);  // h1 @ W2l
    float* agg2  = (float*)take((size_t)NNODES * OUTF * 4);  // scatter_mean(y1)*deg
    float* hh2   = (float*)take((size_t)NNODES * OUTF * 4);
    float* zpart = (float*)take((size_t)MP_CHUNKS * OUTF * 4);
    float* zbuf  = (float*)take((size_t)SEQ * OUTF * 4);
    float* cbuf  = (float*)take((size_t)SEQ * OUTF * 4);

    // ---- prologue: pack weights into bf16 WMMA fragments, zero scalar outs ----
    {
        int n1 = INCH * H2, n2 = H2 * OUTF;
        pack_b_k<<<(n1 + 255) / 256, 256, 0, stream>>>(W1l, W1l_b, INCH, H2);
        pack_b_k<<<(n1 + 255) / 256, 256, 0, stream>>>(W1r, W1r_b, INCH, H2);
        pack_b_k<<<(n2 + 255) / 256, 256, 0, stream>>>(W2l, W2l_b, H2, OUTF);
        pack_b_k<<<(n2 + 255) / 256, 256, 0, stream>>>(W2r, W2r_b, H2, OUTF);
        zero_out2_k<<<1, 32, 0, stream>>>(out);
    }

    // ---- per-timestep pipeline ----
    for (int t = 0; t < SEQ; ++t) {
        const float* xt  = x + (size_t)t * NNODES * INCH;
        const int*   src = ei + (size_t)t * 2 * NEDGES;
        const int*   dst = src + NEDGES;

        // conv1: scatter-sum(x) + dual GEMM (fused mean, bias, ReLU) -> h1
        fill0_k<<<(NNODES * INCH + 255) / 256, 256, 0, stream>>>(agg1, NNODES * INCH);
        fill0_k<<<(NNODES + 255) / 256, 256, 0, stream>>>(deg, NNODES);
        degree_k<<<NEDGES / 256, 256, 0, stream>>>(dst, deg, NEDGES);
        scatter_add_k<<<NEDGES, 128, 0, stream>>>(xt, src, dst, agg1, INCH);
        gemm_dual_wmma_k<<<dim3(H2 / 16, NNODES / 128), 256, 0, stream>>>(
            agg1, xt, W1l_b, W1r_b, deg, b1l, h1, NNODES, H2, INCH);

        // conv2 (aggregation commuted through W2l to halve scatter traffic):
        //   y1 = h1 @ W2l ; agg2 = scatter_sum(y1) ; h2 = agg2/deg + h1@W2r + b2l
        gemm_single_wmma_k<<<dim3(OUTF / 16, NNODES / 128), 256, 0, stream>>>(
            h1, W2l_b, y1, NNODES, OUTF, H2);
        fill0_k<<<(NNODES * OUTF + 255) / 256, 256, 0, stream>>>(agg2, NNODES * OUTF);
        scatter_add_k<<<NEDGES, 256, 0, stream>>>(y1, src, dst, agg2, OUTF);
        gemm_addagg_wmma_k<<<dim3(OUTF / 16, NNODES / 128), 256, 0, stream>>>(
            agg2, h1, W2r_b, deg, b2l, hh2, NNODES, OUTF, H2);

        // global max pool -> z[t] (two-stage, deterministic)
        maxpool_part_k<<<MP_CHUNKS, OUTF, 0, stream>>>(hh2, zpart, NNODES / MP_CHUNKS);
        maxpool_fin_k<<<1, OUTF, 0, stream>>>(zpart, zbuf + (size_t)t * OUTF, MP_CHUNKS);
    }

    // ---- epilogue: GRU cell, CPC loss, hidden copy ----
    gru_k<<<SEQ, OUTF, 0, stream>>>(zbuf, Wih, bih, bhh, cbuf);
    cpc_k<<<dim3(CPC_CNT, CPC_TS), 256, 0, stream>>>(zbuf, cbuf, out);
    copy_hidden_k<<<(HID_LEN * OUTF + 255) / 256, 256, 0, stream>>>(cbuf, out);
}
